// GIN_24893630447616
// MI455X (gfx1250) — matrix operand order repaired and checked by reading, not compiled, not run on hardware
//
#include <hip/hip_runtime.h>

#define NN 50000
#define DD 128
#define NE 600000
#define NG 64
#define DO 10

typedef __attribute__((ext_vector_type(16))) _Float16 v16h;
typedef __attribute__((ext_vector_type(8)))  _Float16 v8h;
typedef __attribute__((ext_vector_type(8)))  float    v8f;

__device__ __forceinline__ void atomAddF(float* p, float v) {
    __hip_atomic_fetch_add(p, v, __ATOMIC_RELAXED, __HIP_MEMORY_SCOPE_AGENT);
}

// ---------------- utility ----------------
__global__ void zero_kernel(float* p, int n) {
    int i = blockIdx.x * blockDim.x + threadIdx.x;
    int stride = gridDim.x * blockDim.x;
    for (; i < n; i += stride) p[i] = 0.f;
}

// ---------------- degrees ----------------
__global__ void degree_kernel(const int* __restrict__ src, const int* __restrict__ dst,
                              float* deg_out, float* deg_in) {
    int e = blockIdx.x * blockDim.x + threadIdx.x;
    if (e >= NE) return;
    atomAddF(&deg_out[src[e]], 1.f);
    atomAddF(&deg_in[dst[e]], 1.f);
}

__global__ void invsqrt_kernel(const float* __restrict__ deg_out,
                               const float* __restrict__ deg_in,
                               float* iso, float* isi) {
    int i = blockIdx.x * blockDim.x + threadIdx.x;
    if (i >= NN) return;
    iso[i] = rsqrtf(fmaxf(deg_out[i], 1.f));
    isi[i] = rsqrtf(fmaxf(deg_in[i], 1.f));
}

// ---------------- weight convert: W[k][n] (f32 row-major) -> WT[n][k] (f16) ----------------
__global__ void convW_kernel(const float* __restrict__ W, _Float16* __restrict__ WT) {
    int idx = blockIdx.x * blockDim.x + threadIdx.x;   // idx = n*128 + k
    if (idx >= DD * DD) return;
    int n = idx >> 7, k = idx & 127;
    WT[idx] = (_Float16)W[k * DD + n];
}

// ---------------- edge scatter: agg[dst] += x[src] * iso[src] ----------------
// one wave (32 lanes) per edge, each lane carries 4 floats
__global__ __launch_bounds__(256) void scatter_kernel(const float* __restrict__ x,
        const int* __restrict__ src, const int* __restrict__ dst,
        const float* __restrict__ iso, float* __restrict__ agg) {
    long tid = (long)blockIdx.x * blockDim.x + threadIdx.x;
    int e = (int)(tid >> 5);
    if (e >= NE) return;
    int lane = (int)(tid & 31);
    int s = src[e], d = dst[e];
    float sc = iso[s];
    float4 v = *((const float4*)(x + (long)s * DD) + lane);
    float* ad = agg + (long)d * DD + lane * 4;
    atomAddF(ad + 0, v.x * sc);
    atomAddF(ad + 1, v.y * sc);
    atomAddF(ad + 2, v.z * sc);
    atomAddF(ad + 3, v.w * sc);
}

// ---------------- agg * isi -> f16 A-matrix ----------------
__global__ void convertX_kernel(const float* __restrict__ agg,
                                const float* __restrict__ isi,
                                _Float16* __restrict__ Xh) {
    long i = (long)blockIdx.x * blockDim.x + threadIdx.x;
    if (i >= (long)NN * DD) return;
    Xh[i] = (_Float16)(agg[i] * isi[i >> 7]);
}

// ---------------- WMMA GEMM: Y = relu(Xh @ W + b), M=50000 (3125 tiles), N=K=128 ----------------
// one wave per 16x16 output tile; 8 waves/block cover all 8 N-tiles of one M-tile
__global__ __launch_bounds__(256) void gemm_wmma_kernel(const _Float16* __restrict__ Xh,
        const _Float16* __restrict__ WT, const float* __restrict__ bias,
        float* __restrict__ Y) {
    int wave  = threadIdx.x >> 5;          // = ntile (0..7)
    int lane  = threadIdx.x & 31;
    int mtile = blockIdx.x;                // 0..3124
    int r = lane & 15, g = lane >> 4;

    const _Float16* arow = Xh + ((long)mtile * 16 + r) * DD;   // A row for this lane
    int col = wave * 16 + r;                                   // B column for this lane
    const _Float16* bcol = WT + (long)col * DD;                // contiguous K (transposed W)

    v8f acc = {};
#pragma unroll
    for (int ks = 0; ks < 4; ks++) {
        int kb = ks * 32;
        // A fragment: lane group g holds K = kb+8g..kb+8g+7 and kb+16+8g..kb+16+8g+7
        v8h a0 = *(const v8h*)(arow + kb + 8 * g);
        v8h a1 = *(const v8h*)(arow + kb + 16 + 8 * g);
        v16h a = __builtin_shufflevector(a0, a1,
                 0, 1, 2, 3, 4, 5, 6, 7, 8, 9, 10, 11, 12, 13, 14, 15);
        // B fragment: lane group g holds K = kb+8g..kb+8g+7 and kb+16+8g..kb+16+8g+7
        v8h b0 = *(const v8h*)(bcol + kb + 8 * g);
        v8h b1 = *(const v8h*)(bcol + kb + 16 + 8 * g);
        v16h b = __builtin_shufflevector(b0, b1,
                 0, 1, 2, 3, 4, 5, 6, 7, 8, 9, 10, 11, 12, 13, 14, 15);
        acc = __builtin_amdgcn_wmma_f32_16x16x32_f16(
                  false, a, false, b, (short)0, acc, false, false);
    }

    // C/D layout: element v -> M = v + 8*g, N = lane&15
    float bv = bias[col];
    float* yb = Y + ((long)mtile * 16 + 8 * g) * DD + col;
#pragma unroll
    for (int v = 0; v < 8; v++) {
        float val = acc[v] + bv;
        yb[(long)v * DD] = fmaxf(val, 0.f);
    }
}

// ---------------- per-graph sum / count ----------------
__global__ void graph_sum_kernel(const float* __restrict__ X, const int* __restrict__ gid,
                                 float* gsum, float* gcnt) {
    long tid = (long)blockIdx.x * blockDim.x + threadIdx.x;
    if (tid >= (long)NN * DD) return;
    int node = (int)(tid >> 7), k = (int)(tid & 127);
    int g = gid[node];
    atomAddF(&gsum[g * DD + k], X[tid]);
    if (k == 0) atomAddF(&gcnt[g], 1.f);
}

// ---------------- classifier: out[g] = (gsum[g]/cnt[g]) @ Wc + bc ----------------
__global__ void classifier_kernel(const float* __restrict__ gsum, const float* __restrict__ gcnt,
                                  const float* __restrict__ Wc, const float* __restrict__ bc,
                                  float* __restrict__ out) {
    __shared__ float hg[DD];
    int g = blockIdx.x;
    int t = threadIdx.x;   // 128 threads
    float c = fmaxf(gcnt[g], 1.f);
    hg[t] = gsum[g * DD + t] / c;
    __syncthreads();
    if (t < DO) {
        float acc = bc[t];
        for (int k = 0; k < DD; k++) acc += hg[k] * Wc[k * DO + t];
        out[g * DO + t] = acc;
    }
}

// ---------------- workspace layout (bytes) ----------------
#define AGG_OFF   ((size_t)0)                        // 50000*128*4 = 25,600,000
#define Y_OFF     ((size_t)25600000)                 // 25,600,000
#define XH_OFF    ((size_t)51200000)                 // 12,800,000 (f16)
#define WT1_OFF   ((size_t)64000000)                 // 32,768 (f16)
#define WT2_OFF   ((size_t)64032768)                 // 32,768 (f16)
#define DEGO_OFF  ((size_t)64065536)                 // 200,000
#define DEGI_OFF  ((size_t)64265536)                 // 200,000
#define ISO_OFF   ((size_t)64465536)                 // 200,000
#define ISI_OFF   ((size_t)64665536)                 // 200,000
#define GSUM_OFF  ((size_t)64865536)                 // 32,768
#define GCNT_OFF  ((size_t)64898304)                 // 256

extern "C" void kernel_launch(void* const* d_in, const int* in_sizes, int n_in,
                              void* d_out, int out_size, void* d_ws, size_t ws_size,
                              hipStream_t stream) {
    const float* feat = (const float*)d_in[0];
    const float* W1   = (const float*)d_in[1];
    const float* b1   = (const float*)d_in[2];
    const float* W2   = (const float*)d_in[3];
    const float* b2   = (const float*)d_in[4];
    const float* Wc   = (const float*)d_in[5];
    const float* bc   = (const float*)d_in[6];
    const int*   src  = (const int*)d_in[7];
    const int*   dst  = (const int*)d_in[8];
    const int*   gid  = (const int*)d_in[9];

    char* ws = (char*)d_ws;
    float*    agg  = (float*)(ws + AGG_OFF);
    float*    Y    = (float*)(ws + Y_OFF);
    _Float16* Xh   = (_Float16*)(ws + XH_OFF);
    _Float16* WT1  = (_Float16*)(ws + WT1_OFF);
    _Float16* WT2  = (_Float16*)(ws + WT2_OFF);
    float*    dego = (float*)(ws + DEGO_OFF);
    float*    degi = (float*)(ws + DEGI_OFF);
    float*    iso  = (float*)(ws + ISO_OFF);
    float*    isi  = (float*)(ws + ISI_OFF);
    float*    gsum = (float*)(ws + GSUM_OFF);
    float*    gcnt = (float*)(ws + GCNT_OFF);

    const int T = 256;
    const long NF = (long)NN * DD;           // 6,400,000

    // zero deg_out+deg_in (contiguous), gsum+gcnt (contiguous), agg
    zero_kernel<<<256, T, 0, stream>>>(dego, 2 * NN);
    zero_kernel<<<64,  T, 0, stream>>>(gsum, NG * DD + NG);
    zero_kernel<<<4096, T, 0, stream>>>(agg, (int)NF);

    // degrees -> inverse sqrt
    degree_kernel<<<(NE + T - 1) / T, T, 0, stream>>>(src, dst, dego, degi);
    invsqrt_kernel<<<(NN + T - 1) / T, T, 0, stream>>>(dego, degi, iso, isi);

    // weights to f16, transposed
    convW_kernel<<<(DD * DD + T - 1) / T, T, 0, stream>>>(W1, WT1);
    convW_kernel<<<(DD * DD + T - 1) / T, T, 0, stream>>>(W2, WT2);

    // ---- layer 1 ----
    scatter_kernel<<<(int)(((long)NE * 32 + T - 1) / T), T, 0, stream>>>(feat, src, dst, iso, agg);
    convertX_kernel<<<(int)((NF + T - 1) / T), T, 0, stream>>>(agg, isi, Xh);
    gemm_wmma_kernel<<<NN / 16, T, 0, stream>>>(Xh, WT1, b1, Y);

    // ---- layer 2 ----
    zero_kernel<<<4096, T, 0, stream>>>(agg, (int)NF);
    scatter_kernel<<<(int)(((long)NE * 32 + T - 1) / T), T, 0, stream>>>(Y, src, dst, iso, agg);
    convertX_kernel<<<(int)((NF + T - 1) / T), T, 0, stream>>>(agg, isi, Xh);
    gemm_wmma_kernel<<<NN / 16, T, 0, stream>>>(Xh, WT2, b2, Y);

    // ---- readout + classifier ----
    graph_sum_kernel<<<(int)((NF + T - 1) / T), T, 0, stream>>>(Y, gid, gsum, gcnt);
    classifier_kernel<<<NG, DD, 0, stream>>>(gsum, gcnt, Wc, bc, (float*)d_out);
}
